// GCN_40020505264478
// MI455X (gfx1250) — matrix operand order repaired and checked by reading, not compiled, not run on hardware
//
#include <hip/hip_runtime.h>
#include <hip/hip_bf16.h>

#define N_NODES 100000
#define N_EDGES 3200000
#define N_GRAPHS 512
#define IN_CH 50
#define HID 256
#define KPAD 64   // IN_CH padded to two K=32 WMMA steps
#define XSTRIDE 52  // f32 LDS staging row stride

typedef __attribute__((ext_vector_type(16))) _Float16 v16h;
typedef __attribute__((ext_vector_type(8)))  _Float16 v8h;
typedef __attribute__((ext_vector_type(8)))  float    v8f;

// --- gfx1250 async load-to-LDS path (guarded: falls back to plain LDS stores) ---
#if defined(__HIP_DEVICE_COMPILE__) && __has_builtin(__builtin_amdgcn_global_load_async_to_lds_b32)
#define HAVE_ASYNC_LDS 1
typedef __attribute__((address_space(1))) int glb_int;   // clang prints as '__device__ int'
typedef __attribute__((address_space(3))) int lds_int;   // clang prints as '__shared__ int'
__device__ inline glb_int* to_glb(const void* p) {
    return (glb_int*)(unsigned long long)p;
}
__device__ inline lds_int* to_lds(void* p) {
    return (lds_int*)(unsigned int)(unsigned long long)p;
}
__device__ inline void wait_async0() {
#if __has_builtin(__builtin_amdgcn_s_wait_asynccnt)
    __builtin_amdgcn_s_wait_asynccnt(0);
#else
    asm volatile("s_wait_asynccnt 0x0" ::: "memory");
#endif
}
#else
#define HAVE_ASYNC_LDS 0
#endif

// ---------------------------------------------------------------------------
// Workspace zeroing (harness poisons ws; we re-init every call -> deterministic)
// ---------------------------------------------------------------------------
__global__ void sage_zero_ws(float* __restrict__ p, int n) {
    int i = blockIdx.x * blockDim.x + threadIdx.x;
    int stride = gridDim.x * blockDim.x;
    for (; i < n; i += stride) p[i] = 0.0f;
}

// ---------------------------------------------------------------------------
// One-time weight prep: build f16 B-fragments for both W_l and W_r in the
// exact per-lane WMMA layout, so the hot kernel loads each fragment with one
// aligned 32-byte vector load and zero converts.
// Layout: wf16[mat][frag = ntile*2+kblock][lane][16 halfs]
// ---------------------------------------------------------------------------
__global__ __launch_bounds__(256)
void sage_prep_weights(const float* __restrict__ Wl, const float* __restrict__ Wr,
                       _Float16* __restrict__ wf16) {
    int tid = blockIdx.x * blockDim.x + threadIdx.x;  // 0..2047
    if (tid >= 2 * 32 * 32) return;
    int mat   = tid >> 10;         // 0: Wl, 1: Wr
    int f     = (tid >> 5) & 31;   // fragment index: ntile*2 + kblock
    int lane  = tid & 31;
    int ntile = f >> 1;
    int kblock = f & 1;
    int n  = ntile * 16 + (lane & 15);
    int kh = (lane >> 4) * 16;
    const float* row = (mat ? Wr : Wl) + n * IN_CH;
    _Float16* dst = wf16 + ((size_t)mat * 32 * 32 + (size_t)f * 32 + lane) * 16;
#pragma unroll
    for (int i = 0; i < 16; ++i) {
        int c = kblock * 32 + kh / 2 + i + (i >= 8 ? 8 : 0);
        dst[i] = (c < IN_CH) ? (_Float16)row[c] : (_Float16)0.0f;
    }
}

// ---------------------------------------------------------------------------
// Edge scatter: agg[dst] += x[src], cnt[dst] += 1.
// 13 channel-quads per edge; float2 loads (rows are 8B aligned: 50*4=200 bytes).
// x (20MB) and agg (20MB) are L2-resident on MI455X (192MB L2), so the random
// gather + atomics resolve in L2, not HBM.
// ---------------------------------------------------------------------------
__global__ __launch_bounds__(256)
void sage_edge_scatter(const int* __restrict__ ei, const float* __restrict__ x,
                       float* __restrict__ agg, float* __restrict__ cnt) {
    long long tid = (long long)blockIdx.x * blockDim.x + threadIdx.x;
    const long long total = (long long)N_EDGES * 13;
    if (tid >= total) return;
    int e = (int)(tid / 13);
    int q = (int)(tid % 13);
    int s = ei[e];
    int d = ei[N_EDGES + e];
    const float* xs = x + (long long)s * IN_CH;
    float* ad = agg + (long long)d * IN_CH;
    __builtin_prefetch(xs, 0, 0);  // global_prefetch_b8
    if (q == 0) atomicAdd(&cnt[d], 1.0f);
    int c0 = q * 4;
    float2 v0 = *(const float2*)(xs + c0);
    atomicAdd(ad + c0 + 0, v0.x);
    atomicAdd(ad + c0 + 1, v0.y);
    if (q < 12) {
        float2 v1 = *(const float2*)(xs + c0 + 2);
        atomicAdd(ad + c0 + 2, v1.x);
        atomicAdd(ad + c0 + 3, v1.y);
    }
}

// ---------------------------------------------------------------------------
// Per-graph node counts
// ---------------------------------------------------------------------------
__global__ void sage_batch_count(const int* __restrict__ batch, float* __restrict__ gcnt) {
    int i = blockIdx.x * blockDim.x + threadIdx.x;
    if (i < N_NODES) atomicAdd(&gcnt[batch[i]], 1.0f);
}

// ---------------------------------------------------------------------------
// WMMA fragment helpers, per CDNA5 ISA 7.12.2 VGPR layouts (wave32).
// A 16x32 f16: lane group 0 (lanes 0-15, M=lane) holds K {0..7, 16..23};
//              lane group 1 (lanes 16-31) holds K {8..15, 24..31}.
// C/D 16x16 f32: VGPR r <-> M = r + 8*(lane>=16), N = lane&15.
// ---------------------------------------------------------------------------
__device__ inline v8f wmma_f16(v16h a, v16h b, v8f c) {
    return __builtin_amdgcn_wmma_f32_16x16x32_f16(
        /*neg_a=*/false, a, /*neg_b=*/false, b,
        /*c_mod=*/(short)0, c, /*reuse_a=*/false, /*reuse_b=*/false);
}

__device__ inline v16h load_A_lds(const _Float16* sA /*16 x KPAD*/, int lane, int kblock) {
    const int m    = lane & 15;
    const int koff = (lane >> 4) * 8;
    const int base = m * KPAD + kblock * 32;
    v8h lo = *(const v8h*)(sA + base + koff);        // K = kb*32 + koff + 0..7
    v8h hi = *(const v8h*)(sA + base + 16 + koff);   // K = kb*32 + 16 + koff + 0..7
    v16h r;
#pragma unroll
    for (int i = 0; i < 8; ++i) { r[i] = lo[i]; r[8 + i] = hi[i]; }
    return r;
}

// ---------------------------------------------------------------------------
// Fused node kernel: h = leaky_relu(mean_agg @ Wl^T + bl + x @ Wr^T) and
// pooled[batch[node]] += h  (h never hits memory: saves 200+MB of traffic).
// Block = 16 nodes x 256 hidden; 8 waves, each owns two 16-wide N-tiles.
// Tiles staged into LDS via gfx1250 async load-to-LDS when available.
// ---------------------------------------------------------------------------
__global__ __launch_bounds__(256)
void sage_node_wmma(const float* __restrict__ x, const float* __restrict__ agg,
                    const float* __restrict__ cnt, const int* __restrict__ batch,
                    const _Float16* __restrict__ wf16, const float* __restrict__ bl,
                    float* __restrict__ pooled) {
    const int mbase = blockIdx.x * 16;
    __shared__ __align__(16) float    sXf[16][XSTRIDE];    // raw f32 staging
    __shared__ __align__(16) float    sAggf[16][XSTRIDE];
    __shared__ __align__(16) _Float16 sAgg[16][KPAD];      // f16 A tiles
    __shared__ __align__(16) _Float16 sX[16][KPAD];
    __shared__ float s_rc[16];
    __shared__ int   s_g[16];

    const int t = threadIdx.x;

    // Stage raw f32 tiles (16 nodes x 50 ch, both x and agg) into LDS.
    for (int e = t; e < 16 * IN_CH; e += 256) {
        int row = e / IN_CH, col = e % IN_CH;
        long long off = (long long)(mbase + row) * IN_CH + col;
#if HAVE_ASYNC_LDS
        __builtin_amdgcn_global_load_async_to_lds_b32(
            to_glb(x + off),   to_lds(&sXf[row][col]),   0, 0);
        __builtin_amdgcn_global_load_async_to_lds_b32(
            to_glb(agg + off), to_lds(&sAggf[row][col]), 0, 0);
#else
        sXf[row][col]   = x[off];
        sAggf[row][col] = agg[off];
#endif
    }
    if (t < 16) {
        int node = mbase + t;
        s_rc[t] = 1.0f / fmaxf(cnt[node], 1.0f);
        s_g[t]  = batch[node];
    }
#if HAVE_ASYNC_LDS
    wait_async0();
#endif
    __syncthreads();

    // Convert to f16 A tiles (fold mean-scaling into agg), zero-pad K to 64.
#pragma unroll
    for (int k = 0; k < 4; ++k) {
        int e   = t * 4 + k;
        int row = e >> 6;
        int col = e & 63;
        float av = 0.0f, xv = 0.0f;
        if (col < IN_CH) {
            av = sAggf[row][col] * s_rc[row];
            xv = sXf[row][col];
        }
        sAgg[row][col] = (_Float16)av;
        sX[row][col]   = (_Float16)xv;
    }
    __syncthreads();

    const int lane = t & 31;
    const int wave = t >> 5;

    // A fragments are shared by both N-tiles handled by this wave.
    v16h aAgg0 = load_A_lds(&sAgg[0][0], lane, 0);
    v16h aAgg1 = load_A_lds(&sAgg[0][0], lane, 1);
    v16h aX0   = load_A_lds(&sX[0][0],   lane, 0);
    v16h aX1   = load_A_lds(&sX[0][0],   lane, 1);

    const _Float16* wl16 = wf16;                 // [32 frag][32 lane][16]
    const _Float16* wr16 = wf16 + 32 * 32 * 16;
    const int mh = (lane >> 4) << 3;  // row offset of this lane group in C/D

#pragma unroll
    for (int nt = 0; nt < 2; ++nt) {
        const int ntile = wave * 2 + nt;
        const int fb = ntile * 2;    // fragment base: kblock 0/1
        const size_t o0 = (size_t)((fb + 0) * 32 + lane) * 16;
        const size_t o1 = (size_t)((fb + 1) * 32 + lane) * 16;
        v16h bl0 = *(const v16h*)(wl16 + o0);
        v16h bl1 = *(const v16h*)(wl16 + o1);
        v16h br0 = *(const v16h*)(wr16 + o0);
        v16h br1 = *(const v16h*)(wr16 + o1);

        v8f acc = {};
        acc = wmma_f16(aAgg0, bl0, acc);   // mean_agg @ Wl^T  (K 0..31)
        acc = wmma_f16(aAgg1, bl1, acc);   //                  (K 32..63, zero-padded)
        acc = wmma_f16(aX0,   br0, acc);   // x @ Wr^T         (K 0..31)
        acc = wmma_f16(aX1,   br1, acc);   //                  (K 32..63)

        const int n = ntile * 16 + (lane & 15);
        const float bias = bl[n];
#pragma unroll
        for (int r = 0; r < 8; ++r) {
            float v = acc[r] + bias;
            v = (v > 0.0f) ? v : 0.01f * v;                // leaky_relu(0.01)
            atomicAdd(&pooled[s_g[r + mh] * HID + n], v);  // graph pooling (L2-resident)
        }
    }
}

// ---------------------------------------------------------------------------
// Head: out[g] = (pooled[g]/count[g]) @ Wc^T + bc   (512 x 2)
// ---------------------------------------------------------------------------
__global__ __launch_bounds__(256)
void sage_pool_head(const float* __restrict__ pooled, const float* __restrict__ gcnt,
                    const float* __restrict__ Wc, const float* __restrict__ bc,
                    float* __restrict__ out) {
    const int g = blockIdx.x;
    const int t = threadIdx.x;  // 256 = HID
    __shared__ float r0[256];
    __shared__ float r1[256];
    const float rg = 1.0f / fmaxf(gcnt[g], 1.0f);
    const float v  = pooled[g * HID + t] * rg;
    r0[t] = v * Wc[t];
    r1[t] = v * Wc[HID + t];
    __syncthreads();
#pragma unroll
    for (int s = 128; s > 0; s >>= 1) {
        if (t < s) { r0[t] += r0[t + s]; r1[t] += r1[t + s]; }
        __syncthreads();
    }
    if (t == 0) {
        out[g * 2 + 0] = r0[0] + bc[0];
        out[g * 2 + 1] = r1[0] + bc[1];
    }
}

// ---------------------------------------------------------------------------
extern "C" void kernel_launch(void* const* d_in, const int* in_sizes, int n_in,
                              void* d_out, int out_size, void* d_ws, size_t ws_size,
                              hipStream_t stream) {
    const float* x     = (const float*)d_in[0];
    const int*   ei    = (const int*)d_in[1];
    const int*   batch = (const int*)d_in[2];
    const float* Wl    = (const float*)d_in[3];
    const float* bl    = (const float*)d_in[4];
    const float* Wr    = (const float*)d_in[5];
    const float* Wc    = (const float*)d_in[6];
    const float* bc    = (const float*)d_in[7];
    float* out = (float*)d_out;

    // Workspace carve: agg | cnt | pooled | gcnt (f32, ~21 MB) | wf16 (64 KB)
    float* agg    = (float*)d_ws;
    float* cnt    = agg + (size_t)N_NODES * IN_CH;
    float* pooled = cnt + N_NODES;
    float* gcnt   = pooled + (size_t)N_GRAPHS * HID;
    _Float16* wf16 = (_Float16*)(gcnt + N_GRAPHS);   // 2*32*32*16 halfs, 32B-aligned

    const int zn = N_NODES * IN_CH + N_NODES + N_GRAPHS * HID + N_GRAPHS;
    sage_zero_ws<<<(zn + 255) / 256, 256, 0, stream>>>((float*)d_ws, zn);

    sage_prep_weights<<<8, 256, 0, stream>>>(Wl, Wr, wf16);

    const long long ethreads = (long long)N_EDGES * 13;
    sage_edge_scatter<<<(int)((ethreads + 255) / 256), 256, 0, stream>>>(ei, x, agg, cnt);

    sage_batch_count<<<(N_NODES + 255) / 256, 256, 0, stream>>>(batch, gcnt);

    sage_node_wmma<<<N_NODES / 16, 256, 0, stream>>>(x, agg, cnt, batch, wf16, bl, pooled);

    sage_pool_head<<<N_GRAPHS, 256, 0, stream>>>(pooled, gcnt, Wc, bc, out);
}
